// RBFolution_90872918049493
// MI455X (gfx1250) — compile-verified, matrix operand order checked by reading, not run on hardware
//
#include <hip/hip_runtime.h>
#include <hip/hip_bf16.h>
#include <stdint.h>

typedef __attribute__((ext_vector_type(16))) __bf16 v16bf;
typedef __attribute__((ext_vector_type(8)))  float  v8f;

constexpr int Bn = 32, H = 112, W = 112, C = 32;
constexpr int HO = 110, WO = 110, F = 128, D = 288;   // patch dim = 3*3*32
constexpr int MROWS  = Bn * HO * WO;                  // 387200 (divisible by 16)
constexpr int MTILES = MROWS / 16;                    // 24200 = 3025 blocks * 8 waves
constexpr int KSTEPS = D / 32;                        // 9
constexpr int NTILES = F / 16;                        // 8
constexpr int NFRAGS = KSTEPS * NTILES;               // 72 B fragments
// LDS layout: [lane][frag][8 dwords], padded lane stride for bank spread
constexpr int LANE_STRIDE_DW = NFRAGS * 8 + 4;        // 580 dw; 580%64=4 -> 2-way max conflict
constexpr int BFRAG_DW = 32 * LANE_STRIDE_DW;         // 18560 dw = 74240 B

// round-half-up bf16 pack: 2x v_add + 1x v_perm_b32 (splice high halves)
__device__ __forceinline__ uint32_t pack_bf16(float lo, float hi) {
  uint32_t a = __float_as_uint(lo) + 0x8000u;
  uint32_t b = __float_as_uint(hi) + 0x8000u;
  // result bytes [3,2,1,0] = [b.3, b.2, a.3, a.2]
  return __builtin_amdgcn_perm(b, a, 0x07060302u);
}

union Frag { v16bf v; uint32_t u[8]; uint4 q[2]; };

__global__ __launch_bounds__(256) void rbfolution_wmma_kernel(
    const float* __restrict__ x, const float* __restrict__ ccs,
    const float* __restrict__ beta, float* __restrict__ out) {
  __shared__ __align__(16) uint32_t s_bfrag[BFRAG_DW];
  __shared__ float s_csq[F];
  __shared__ float s_beta[F];

  const int tid = threadIdx.x;

  // ---- one-time staging: ccs -> bf16 B fragments (WMMA register image) ----
  for (int idx = tid; idx < 32 * NFRAGS * 8; idx += 256) {
    int v = idx & 7;
    int rest = idx >> 3;
    int f = rest % NFRAGS;           // f = s*8 + t
    int l = rest / NFRAGS;
    int s = f >> 3;
    int t = f & 7;
    int khalf = (l & 16) ? 8 : 0;    // lanes 16..31 hold K+8 half
    int kq    = (v >= 4) ? 16 : 0;   // VGPRs 4..7 hold K+16 quarter
    int k = 32 * s + khalf + kq + 2 * (v & 3);
    int n = t * 16 + (l & 15);
    s_bfrag[l * LANE_STRIDE_DW + f * 8 + v] =
        pack_bf16(ccs[k * F + n], ccs[(k + 1) * F + n]);
  }
  // ||c||^2 per filter (exact f32) and beta
  if (tid < F) {
    float sum = 0.f;
    for (int k = 0; k < D; ++k) { float c = ccs[k * F + tid]; sum += c * c; }
    s_csq[tid] = sum;
  } else if (tid < 2 * F) {
    s_beta[tid - F] = beta[tid - F];
  }
  __syncthreads();

  const int lane  = tid & 31;
  const int wave  = tid >> 5;
  const int col   = lane & 15;
  const int khalf = (lane & 16) ? 8 : 0;   // A-fragment K-half for this lane
  const int rbase = (lane & 16) >> 1;      // rows 0..7 or 8..15 of C/D tile

  const int tile = blockIdx.x * 8 + wave;  // exactly MTILES waves launched

  // this lane's im2col row (lanes l and l+16 share row l&15)
  int row = tile * 16 + col;
  int b   = row / (HO * WO);
  int rem = row - b * (HO * WO);
  int ho  = rem / WO;
  int wo  = rem - ho * WO;
  const float* rbp = x + (((b * H) + ho) * W + wo) * C;

  const uint32_t* bbase = &s_bfrag[lane * LANE_STRIDE_DW];

  v8f acc[NTILES] = {};
  v8f gacc = {};                     // Gram accumulator: diag = ||p||^2

#pragma unroll
  for (int s = 0; s < KSTEPS; ++s) {
    // k-step s == one contiguous 32-float channel run at (i,j) = (s/3, s%3)
    const float* seg = rbp + ((s / 3) * W + (s % 3)) * C + khalf;
    float4 f0 = *(const float4*)(seg);        // ksub khalf+0..3
    float4 f1 = *(const float4*)(seg + 4);    // ksub khalf+4..7
    float4 f2 = *(const float4*)(seg + 16);   // ksub khalf+16..19
    float4 f3 = *(const float4*)(seg + 20);   // ksub khalf+20..23

    Frag a;
    a.u[0] = pack_bf16(f0.x, f0.y);
    a.u[1] = pack_bf16(f0.z, f0.w);
    a.u[2] = pack_bf16(f1.x, f1.y);
    a.u[3] = pack_bf16(f1.z, f1.w);
    a.u[4] = pack_bf16(f2.x, f2.y);
    a.u[5] = pack_bf16(f2.z, f2.w);
    a.u[6] = pack_bf16(f3.x, f3.y);
    a.u[7] = pack_bf16(f3.z, f3.w);

    // A-fragment layout (lane<->M) == B-fragment layout (lane<->N):
    // wmma(a, a) accumulates the patch Gram matrix; diagonal = ||p||^2.
    gacc = __builtin_amdgcn_wmma_f32_16x16x32_bf16(
        false, a.v, false, a.v, (short)0, gacc, false, false);

#pragma unroll
    for (int t = 0; t < NTILES; ++t) {
      Frag bf;
      bf.q[0] = *(const uint4*)(bbase + (s * 8 + t) * 8);
      bf.q[1] = *(const uint4*)(bbase + (s * 8 + t) * 8 + 4);
      acc[t] = __builtin_amdgcn_wmma_f32_16x16x32_bf16(
          false, a.v, false, bf.v, (short)0, acc[t], false, false);
    }
  }

  // gather ||p||^2 for the 8 rows this lane covers:
  // diag of row m sits in gacc[m&7] at lane (m<8 ? m : m+16)
  const int diagOfs = (lane & 16) ? 24 : 0;
  float psqv[8];
#pragma unroll
  for (int r = 0; r < 8; ++r) psqv[r] = __shfl(gacc[r], r + diagOfs, 32);

  // epilogue: out = exp(-beta * (||p||^2 - 2*(p.c) + ||c||^2))
  float* orow = out + (size_t)(tile * 16 + rbase) * F + col;
#pragma unroll
  for (int t = 0; t < NTILES; ++t) {
    float cs = s_csq[t * 16 + col];
    float bt = s_beta[t * 16 + col];
#pragma unroll
    for (int r = 0; r < 8; ++r) {
      float score = psqv[r] - 2.0f * acc[t][r] + cs;
      orow[(size_t)r * F + t * 16] = __expf(-bt * score);
    }
  }
}

extern "C" void kernel_launch(void* const* d_in, const int* in_sizes, int n_in,
                              void* d_out, int out_size, void* d_ws, size_t ws_size,
                              hipStream_t stream) {
  const float* x    = (const float*)d_in[0];
  const float* ccs  = (const float*)d_in[1];
  const float* beta = (const float*)d_in[2];
  float* out = (float*)d_out;
  (void)in_sizes; (void)n_in; (void)out_size; (void)d_ws; (void)ws_size;
  // 3025 blocks * 8 waves = 24200 waves == MTILES (one 16x128 tile per wave)
  rbfolution_wmma_kernel<<<dim3(MTILES / 8), dim3(256), 0, stream>>>(x, ccs, beta, out);
}
